// OptimizedConv1DSelfONN_22694607192641
// MI455X (gfx1250) — matrix operand order repaired and checked by reading, not compiled, not run on hardware
//
#include <hip/hip_runtime.h>
#include <hip/hip_bf16.h>

typedef __attribute__((ext_vector_type(16))) _Float16 v16h;
typedef __attribute__((ext_vector_type(8)))  _Float16 v8h;
typedef __attribute__((ext_vector_type(4)))  _Float16 v4h;
typedef __attribute__((ext_vector_type(8)))  float    v8f;

// Problem dims (fixed by the reference)
constexpr int B_N   = 16;
constexpr int CIN   = 64;
constexpr int T_LEN = 16384;
constexpr int COUT  = 64;
constexpr int KW    = 9;
constexpr int QB    = 5;
constexpr int PAD   = 4;

constexpr int M_TOT = QB * COUT;     // 320 fused output rows (q*64 + o)
constexpr int KDIM  = CIN * KW;      // 576 reduction length
constexpr int KSTEP = KDIM / 32;     // 18 WMMA k-steps (K=32 each)

// Tiling
constexpr int NT          = 128;           // t-columns per workgroup
constexpr int LDSB_STRIDE = NT + 8;        // 136 halfs/row -> 272B rows, 16B aligned
constexpr int NTHREADS    = 320;           // 10 waves
// Wave decomposition: mgroup = wave%5 -> M rows [64*mgroup, 64*mgroup+64) (one q);
//                     nhalf  = wave/5 -> t-columns [64*nhalf, 64*nhalf+64).

constexpr size_t WPACK_HALFS = (size_t)M_TOT * KDIM;   // 184320 halfs
constexpr size_t WPACK_BYTES = WPACK_HALFS * 2;        // 368640 B in d_ws
constexpr size_t SMEM_BYTES  =
    (size_t)KDIM * LDSB_STRIDE * 2 +   // im2col f16 tile: 156,672 B
    (size_t)COUT * NT * 4;             // f32 output tile:  32,768 B  (total 189,440 B)

// Per-k-step byte strides for running-offset addressing (fit in 32 bits)
constexpr unsigned A_KK_STRIDE = 32u * 16u * 2u;                  // 1024 B
constexpr unsigned B_KK_STRIDE = 32u * (unsigned)LDSB_STRIDE * 2; // 8704 B

// ---------------------------------------------------------------------------
// Prep 1: repack weights into per-lane WMMA A-fragments (f16).
// A-frag layout (16-bit A 16x32, ISA 7.12.2): lane l holds row M=l%16,
// VGPR j holds K = (j<4?0:16) + 8*(l/16) + 2*(j%4) + {0,1}.
// Pack index: (((m_tile*18 + kk)*32 + lane)*16 + h), h = 2j+p.
// ---------------------------------------------------------------------------
__global__ void pack_weights_kernel(const float* __restrict__ cw,
                                    _Float16* __restrict__ wpack) {
    const size_t idx = (size_t)blockIdx.x * blockDim.x + threadIdx.x;
    if (idx >= WPACK_HALFS) return;
    const int h    = (int)(idx & 15);
    const int lane = (int)((idx >> 4) & 31);
    const int kk   = (int)((idx >> 9) % KSTEP);
    const int m    = (int)(idx / (16u * 32u * KSTEP));   // M-tile 0..19
    const int j = h >> 1, p = h & 1, g = lane >> 4;
    const int kloc = ((j < 4) ? 0 : 16) + 8 * g + 2 * (j & 3) + p;
    const int r    = kk * 32 + kloc;                     // K-dim index 0..575
    const int Mrow = m * 16 + (lane & 15);               // 0..319
    const int q = Mrow / COUT, o = Mrow % COUT;
    const int ic = r / KW, k = r % KW;
    // conv_weights layout: (Cout, Cin, K, Q)
    const float w = cw[(((size_t)o * CIN + ic) * KW + k) * QB + q];
    wpack[idx] = (_Float16)w;
}

// ---------------------------------------------------------------------------
// Prep 2: softmax(operator_probs, axis=1) -> opw[o*5+q]
// ---------------------------------------------------------------------------
__global__ void softmax_probs_kernel(const float* __restrict__ probs,
                                     float* __restrict__ opw) {
    const int o = threadIdx.x;
    if (o >= COUT) return;
    float p[QB], mx = -1e30f;
#pragma unroll
    for (int q = 0; q < QB; ++q) { p[q] = probs[o * QB + q]; mx = fmaxf(mx, p[q]); }
    float s = 0.0f;
#pragma unroll
    for (int q = 0; q < QB; ++q) { p[q] = __expf(p[q] - mx); s += p[q]; }
    const float inv = 1.0f / s;
#pragma unroll
    for (int q = 0; q < QB; ++q) opw[o * QB + q] = p[q] * inv;
}

// ---------------------------------------------------------------------------
// Main fused kernel: im2col -> WMMA f16 GEMM -> per-q nonlinearity ->
// softmax-weighted q-reduction (LDS atomics) -> clipped f32 store.
// ---------------------------------------------------------------------------
__global__ __launch_bounds__(NTHREADS, 1)
void conv_onn_main(const float* __restrict__ x,
                   const _Float16* __restrict__ wpack,
                   const float* __restrict__ opw,
                   float* __restrict__ out) {
    extern __shared__ char smem[];
    _Float16* lds_b   = (_Float16*)smem;                                  // [576][136]
    float*    lds_out = (float*)(smem + (size_t)KDIM * LDSB_STRIDE * 2);  // [64][128]

    const int tid = threadIdx.x;
    const int b   = blockIdx.y;
    const int t0  = blockIdx.x * NT;
    const float* xb = x + (size_t)b * CIN * T_LEN;

    // Zero the q-reduction tile
    for (int i = tid; i < COUT * NT; i += NTHREADS) lds_out[i] = 0.0f;

    // ---- Stage im2col tile (f16, clipped input) into LDS ----
    // Row r = ic*9+k holds x[b][ic][t0-4+k + n], n = 0..127 (+zeros OOB).
    constexpr int NG = NT / 4;
    for (int job = tid; job < KDIM * NG; job += NTHREADS) {
        const int r  = job / NG;
        const int n  = (job % NG) * 4;
        const int ic = r / KW, k = r % KW;
        const int t  = t0 - PAD + k + n;
        const float* src = xb + (size_t)ic * T_LEN + t;
        v4h hv;
#pragma unroll
        for (int e = 0; e < 4; ++e) {
            const int tt = t + e;
            float v = (tt >= 0 && tt < T_LEN) ? src[e] : 0.0f;
            v = fminf(fmaxf(v, -3.0f), 3.0f);
            hv[e] = (_Float16)v;
        }
        *(v4h*)&lds_b[(size_t)r * LDSB_STRIDE + n] = hv;   // 8B-aligned ds_store_b64
    }
    __syncthreads();

    // ---- WMMA main loop: 4 M-tiles x 4 N-subtiles per wave ----
    const int wave   = tid >> 5;
    const int lane   = tid & 31;
    const int mgroup = wave % 5;        // q branch; M rows [64*mgroup, +64)
    const int nhalf  = wave / 5;        // t-columns [64*nhalf, +64)
    const int q      = mgroup;

    v8f zero = {};
    v8f acc[4][4];
#pragma unroll
    for (int mt = 0; mt < 4; ++mt)
#pragma unroll
        for (int ns = 0; ns < 4; ++ns) acc[mt][ns] = zero;

    // Running 32-bit byte offsets (constant stride per k-step) so the backend
    // can keep saddr + 32-bit voffset addressing with one v_add per fragment.
    const char* wbase = (const char*)wpack;
    const char* bbase = (const char*)lds_b;
    unsigned aoff[4];
#pragma unroll
    for (int mt = 0; mt < 4; ++mt) {
        const unsigned mtile = (unsigned)(mgroup * 4 + mt);
        aoff[mt] = ((mtile * KSTEP * 32u) + (unsigned)lane) * 32u;  // bytes
    }
    unsigned boff = (unsigned)(lane & 15) * 2u + (unsigned)nhalf * 128u;

#pragma unroll 2
    for (int kk = 0; kk < KSTEP; ++kk) {
        // A-fragments for the 4 M-tiles (clause-grouped global_load_b128, L2-hot)
        v16h afrag[4];
#pragma unroll
        for (int mt = 0; mt < 4; ++mt) {
            const v8h* ap = (const v8h*)(wbase + aoff[mt]);
            afrag[mt] = __builtin_shufflevector(ap[0], ap[1],
                0, 1, 2, 3, 4, 5, 6, 7, 8, 9, 10, 11, 12, 13, 14, 15);
            aoff[mt] += A_KK_STRIDE;
        }
        // B-fragments: lane = N column (lane&15), 16 K rows in table order.
#pragma unroll
        for (int ns = 0; ns < 4; ++ns) {
            const _Float16* bp = (const _Float16*)(bbase + boff) + ns * 16;
            v16h bfrag;
#pragma unroll
            for (int s = 0; s < 16; ++s)
                bfrag[s] = bp[((s & 7) + 8 * (lane >> 4) + ((s >> 3) << 4)) * LDSB_STRIDE];
#pragma unroll
            for (int mt = 0; mt < 4; ++mt) {   // 4 WMMAs per LDS round-trip
                acc[mt][ns] = __builtin_amdgcn_wmma_f32_16x16x32_f16(
                    false, afrag[mt], false, bfrag,
                    (short)0, acc[mt][ns], false, false);
            }
        }
        boff += B_KK_STRIDE;
    }

    // ---- Epilogue: clip, nonlinearity(q), weight, reduce over q via LDS ----
    const int hi  = lane >> 4;   // D-frag: VGPR j -> M = j + 8*hi, N = lane&15
    const int nlo = lane & 15;
    float wq[4][8];
#pragma unroll
    for (int mt = 0; mt < 4; ++mt)
#pragma unroll
        for (int j = 0; j < 8; ++j) {
            const int o = mt * 16 + j + 8 * hi;   // 0..63 (full channel range)
            wq[mt][j] = opw[o * QB + q];
        }

    // Unswitch the wave-uniform q branch outside the unrolled loops.
    auto epilog = [&](auto act) {
#pragma unroll
        for (int mt = 0; mt < 4; ++mt) {
#pragma unroll
            for (int ns = 0; ns < 4; ++ns) {
#pragma unroll
                for (int j = 0; j < 8; ++j) {
                    float v = acc[mt][ns][j];
                    v = fminf(fmaxf(v, -8.0f), 8.0f);
                    const float a = act(v);
                    const int o = mt * 16 + j + 8 * hi;
                    const int n = nhalf * 64 + ns * 16 + nlo;
                    atomicAdd(&lds_out[o * NT + n], a * wq[mt][j]);  // ds_add_f32
                }
            }
        }
    };
    if      (q == 0) epilog([] __device__ (float v) { return v; });
    else if (q == 1) epilog([] __device__ (float v) { return __sinf(v); });
    else if (q == 2) epilog([] __device__ (float v) { return __cosf(v); });
    else if (q == 3) epilog([] __device__ (float v) { return tanhf(v); });
    else             epilog([] __device__ (float v) {
                         return __expf(fminf(fmaxf(v, -5.0f), 5.0f)); });
    __syncthreads();

    // ---- Clipped coalesced writeout ----
    float* ob = out + (size_t)b * COUT * T_LEN + t0;
    for (int idx = tid; idx < COUT * NT; idx += NTHREADS) {
        const int o = idx / NT, n = idx % NT;
        float v = lds_out[idx];
        v = fminf(fmaxf(v, -5.0f), 5.0f);
        ob[(size_t)o * T_LEN + n] = v;
    }
}

// ---------------------------------------------------------------------------
extern "C" void kernel_launch(void* const* d_in, const int* in_sizes, int n_in,
                              void* d_out, int out_size, void* d_ws, size_t ws_size,
                              hipStream_t stream) {
    const float* x  = (const float*)d_in[0];   // (16, 64, 16384) f32
    const float* cw = (const float*)d_in[1];   // (64, 64, 9, 5)  f32
    const float* op = (const float*)d_in[2];   // (64, 5)         f32
    float* out = (float*)d_out;                // (16, 64, 16384) f32

    _Float16* wpack = (_Float16*)d_ws;
    float*    opw   = (float*)((char*)d_ws + WPACK_BYTES);

    pack_weights_kernel<<<(int)((WPACK_HALFS + 255) / 256), 256, 0, stream>>>(cw, wpack);
    softmax_probs_kernel<<<1, 64, 0, stream>>>(op, opw);

    dim3 grid(T_LEN / NT, B_N);
    conv_onn_main<<<grid, NTHREADS, SMEM_BYTES, stream>>>(x, wpack, opw, out);
}
